// SEGandDETMeasure_27685359190288
// MI455X (gfx1250) — compile-verified, matrix operand order checked by reading, not run on hardware
//
#include <hip/hip_runtime.h>

// SEG/DET measure for label masks (labels 0..255, 0 = background).
//
// MI455X reasoning:
//  - Dominant cost: streaming 2 x 64MB int32 label maps once  -> ~5.8us at 23.3 TB/s.
//  - Confusion matrix is 256x256 u32 = 256KB. CDNA5 WGP has 320KB LDS, so the FULL
//    histogram is privatized per workgroup in LDS (ds_add_u32, random bins -> low
//    conflict), merged once with global atomics. Global atomic traffic drops from
//    16.7M pixel-atomics to ~ (numWG x nonzero-bins) one-time merges.
//  - Pixel loop unrolled 2x: LOADcnt and DScnt are independent CDNA5 counters, so
//    4 b128 loads stay in flight past the dependent LDS atomics (more MLP per wave);
//    global_prefetch_b8 issued one unroll ahead (speculative, OOB silently dropped).
//  - Boolean overlap-matrix column sums (needed for `pairs` and `ea`) are computed
//    with V_WMMA_I32_16X16X64_IU8: A = all-ones 16x64 u8, B = 64x16 overlap tile
//    from LDS (transposed storage -> two contiguous b128 LDS reads per K-tile),
//    4 chained WMMAs accumulate the full 256-row sum per 16-column tile.
//  - The 255-step greedy matching is inherently sequential and tiny: one thread.

#define LBL 256
#define NBINS (LBL * LBL)

typedef __attribute__((ext_vector_type(8))) int v8i;

__global__ void zero_hist_kernel(unsigned int* __restrict__ C) {
  int i = blockIdx.x * blockDim.x + threadIdx.x;   // covers NBINS/4 uint4
  ((uint4*)C)[i] = make_uint4(0u, 0u, 0u, 0u);
}

__global__ __launch_bounds__(1024) void hist_kernel(const int4* __restrict__ pred4,
                                                    const int4* __restrict__ gt4,
                                                    unsigned int* __restrict__ C,
                                                    int n4) {
  // Full 256KB per-workgroup histogram in CDNA5's 320KB LDS.
  __shared__ unsigned int h[NBINS];
  for (int b = threadIdx.x; b < NBINS; b += blockDim.x) h[b] = 0u;
  __syncthreads();

  const int stride = gridDim.x * blockDim.x;
  int i = blockIdx.x * blockDim.x + threadIdx.x;

  // 2x-unrolled stream: 4 outstanding global_load_b128 before the ds atomics.
  for (; i + stride < n4; i += 2 * stride) {
    int4 p0 = pred4[i];
    int4 g0 = gt4[i];
    int4 p1 = pred4[i + stride];
    int4 g1 = gt4[i + stride];
    __builtin_prefetch(&pred4[i + 2 * stride], 0, 1);  // global_prefetch_b8
    __builtin_prefetch(&gt4[i + 2 * stride], 0, 1);
    atomicAdd(&h[(g0.x << 8) | p0.x], 1u);   // ds_add_u32
    atomicAdd(&h[(g0.y << 8) | p0.y], 1u);
    atomicAdd(&h[(g0.z << 8) | p0.z], 1u);
    atomicAdd(&h[(g0.w << 8) | p0.w], 1u);
    atomicAdd(&h[(g1.x << 8) | p1.x], 1u);
    atomicAdd(&h[(g1.y << 8) | p1.y], 1u);
    atomicAdd(&h[(g1.z << 8) | p1.z], 1u);
    atomicAdd(&h[(g1.w << 8) | p1.w], 1u);
  }
  if (i < n4) {
    int4 p = pred4[i];
    int4 g = gt4[i];
    atomicAdd(&h[(g.x << 8) | p.x], 1u);
    atomicAdd(&h[(g.y << 8) | p.y], 1u);
    atomicAdd(&h[(g.z << 8) | p.z], 1u);
    atomicAdd(&h[(g.w << 8) | p.w], 1u);
  }
  __syncthreads();

  for (int b = threadIdx.x; b < NBINS; b += blockDim.x) {
    unsigned int v = h[b];
    if (v) atomicAdd(&C[b], v);            // global_atomic_add_u32 merge
  }
}

__global__ __launch_bounds__(256) void stats_kernel(const unsigned int* __restrict__ C,
                                                    float* __restrict__ out) {
  __shared__ __align__(16) unsigned char ovT[LBL][LBL];  // overlap, transposed: ovT[p][g]
  __shared__ unsigned int gt_size[LBL];
  __shared__ unsigned int pred_size[LBL];
  __shared__ unsigned int inter[LBL];
  __shared__ unsigned int colsum[LBL];     // overlap.sum(axis=0) per pred label
  __shared__ int best_p[LBL];
  __shared__ unsigned char has[LBL];
  __shared__ unsigned char used[LBL];

  const int t = threadIdx.x;   // 0..255 : one label per thread

  // ---- Row pass: gt_size + majority-match scan (Cnz excludes row 0 / col 0) ----
  unsigned int gtsz = 0;
  for (int p = 0; p < LBL; ++p) gtsz += C[t * LBL + p];
  gt_size[t] = gtsz;

  int bp = 0;
  unsigned int itc = 0;
  unsigned char hv = 0;
  if (t > 0) {
    for (int p = 1; p < LBL; ++p) {
      unsigned int c = C[t * LBL + p];
      // 2*c > gt_size: exact int compare; c <= 2^24 so 2*c fits u32.
      if (!hv && (2u * c > gtsz)) { hv = 1; bp = p; itc = c; }   // first (unique) true
    }
  }
  best_p[t] = bp; inter[t] = itc; has[t] = hv;

  // ---- Column pass: pred_size + transposed overlap bytes ----
  unsigned int psz = 0;
  for (int g = 0; g < LBL; ++g) {
    unsigned int c = C[g * LBL + t];
    psz += c;
    ovT[t][g] = (t > 0 && g > 0 && c > 0) ? (unsigned char)1 : (unsigned char)0;
  }
  pred_size[t] = psz;
  used[t] = 0;
  __syncthreads();

  // ---- Wave 0: WMMA column sums of the boolean overlap matrix ----
  // colsum[p] = sum_g ovT[p][g].  A = ones(16x64 u8), B = 64x16 overlap tile.
  // B lane layout (iu8, wave32): lane L -> column n = L&15; K-half = L>>4;
  //   v[0..3] = K (khalf*16 .. khalf*16+15), v[4..7] = K+32.  D rows all equal colsum.
  if (t < 32) {
    const int lane  = t;
    const int nn    = lane & 15;
    const int khalf = lane >> 4;
    v8i a;
#pragma unroll
    for (int i = 0; i < 8; ++i) a[i] = 0x01010101;

    for (int tn = 0; tn < 16; ++tn) {
      const int p = tn * 16 + nn;
      v8i acc = {};
#pragma unroll
      for (int kt = 0; kt < 4; ++kt) {
        const int4 lo = *(const int4*)&ovT[p][kt * 64 + khalf * 16];        // ds_load_b128
        const int4 hi = *(const int4*)&ovT[p][kt * 64 + 32 + khalf * 16];   // ds_load_b128
        v8i b;
        b[0] = lo.x; b[1] = lo.y; b[2] = lo.z; b[3] = lo.w;
        b[4] = hi.x; b[5] = hi.y; b[6] = hi.z; b[7] = hi.w;
        acc = __builtin_amdgcn_wmma_i32_16x16x64_iu8(false, a, false, b, acc,
                                                     false, false);
      }
      if (lane < 16) colsum[p] = (unsigned int)acc[0];
    }
  }
  __syncthreads();

  // ---- Serial tail: reductions + greedy matching (255 steps, inherently sequential) ----
  if (t == 0) {
    int num_gt = 0, num_pred = 0, ea = 0;
    unsigned int pairs = 0;
    for (int l = 1; l < LBL; ++l) {
      if (gt_size[l] > 0) num_gt++;
      if (pred_size[l] > 0) num_pred++;
      pairs += colsum[l];
      if (colsum[l] > 1) ea++;
    }

    int tp = 0;
    float seg_sum = 0.0f;
    for (int gl = 1; gl < LBL; ++gl) {
      int pl = best_p[gl];
      bool ok = (has[gl] != 0) && (used[pl] == 0);
      if (ok) {
        unsigned int uni = gt_size[gl] + pred_size[pl] - inter[gl];
        if (uni < 1u) uni = 1u;
        seg_sum += (float)inter[gl] / (float)uni;
        used[pl] = 1;
        tp++;
      }
    }

    int ngi = num_gt > 0 ? num_gt : 1;
    float ng = (float)ngi;
    float seg = seg_sum / ng;
    int ns = (int)pairs - tp;
    int fn = num_gt - tp;
    int fp = num_pred - tp;
    float det = 1.0f - (float)(fp + fn + ns + ea) / ng;

    bool both_empty = (num_gt == 0) && (num_pred == 0);
    bool any_empty  = (num_gt == 0) || (num_pred == 0);
    if (both_empty)      { seg = 1.0f; det = 1.0f; }
    else if (any_empty)  { seg = 0.0f; det = 0.0f; }

    out[0] = seg;
    out[1] = det;
  }
}

extern "C" void kernel_launch(void* const* d_in, const int* in_sizes, int n_in,
                              void* d_out, int out_size, void* d_ws, size_t ws_size,
                              hipStream_t stream) {
  const int4* pred4 = (const int4*)d_in[0];
  const int4* gt4   = (const int4*)d_in[1];
  unsigned int* C   = (unsigned int*)d_ws;   // 256 KB confusion matrix
  float* out        = (float*)d_out;         // {seg, det}

  const int n  = in_sizes[0];                // 4096*4096
  const int n4 = n >> 2;

  zero_hist_kernel<<<NBINS / 4 / 256, 256, 0, stream>>>(C);
  hist_kernel<<<128, 1024, 0, stream>>>(pred4, gt4, C, n4);
  stats_kernel<<<1, 256, 0, stream>>>(C, out);
}
